// GenSupConLoss_1056561955282
// MI455X (gfx1250) — matrix operand order repaired and checked by reading, not compiled, not run on hardware
//
#include <hip/hip_runtime.h>
#include <hip/hip_bf16.h>

typedef __attribute__((ext_vector_type(16))) _Float16 v16h;
typedef __attribute__((ext_vector_type(8)))  _Float16 v8h;
typedef __attribute__((ext_vector_type(8)))  float    v8f;

#define SC_TEMP 0.07f

__global__ void zero_out_kernel(float* out) { out[0] = 0.0f; }

// Convert concat([f1;f2]) (row-major f32, 2N x D) into f16.
__global__ void prep_features(const float* __restrict__ f1, const float* __restrict__ f2,
                              _Float16* __restrict__ F16, int N, int D) {
    int idx = blockIdx.x * blockDim.x + threadIdx.x;
    int total = 2 * N * D;
    if (idx >= total) return;
    int row = idx / D;
    int col = idx - row * D;
    float v = (row < N) ? f1[row * D + col] : f2[(row - N) * D + col];
    F16[idx] = (_Float16)v;
}

// Build zero-padded (K=32) *unit-normalized* f16 label matrix.
// mask_ij = (l_i/|l_i|) . (l_j/|l_j|) comes straight out of the label WMMA.
__global__ void prep_labels(const float* __restrict__ l1, const float* __restrict__ l2,
                            _Float16* __restrict__ Lp, int N, int C) {
    int row = blockIdx.x * blockDim.x + threadIdx.x;
    if (row >= 2 * N) return;
    const float* src = (row < N) ? (l1 + (size_t)row * C) : (l2 + (size_t)(row - N) * C);
    float s = 0.0f;
    for (int c = 0; c < C; ++c) { float v = src[c]; s += v * v; }
    const float inv = 1.0f / sqrtf(s);       // once per row, off the hot path
    for (int c = 0; c < 32; ++c)
        Lp[row * 32 + c] = (_Float16)((c < C) ? src[c] * inv : 0.0f);
}

__device__ __forceinline__ v16h pack16(v8h lo, v8h hi) {
    return __builtin_shufflevector(lo, hi, 0, 1, 2, 3, 4, 5, 6, 7,
                                           8, 9, 10, 11, 12, 13, 14, 15);
}

// A fragment (16x32 f16): lane L<16 -> row L, K{0..7}+{16..23};
// lane L+16 -> row L, K{8..15}+{24..31}.
__device__ __forceinline__ v16h load_a_frag(const _Float16* base, int lhalf) {
    v8h lo = *reinterpret_cast<const v8h*>(base + lhalf * 8);
    v8h hi = *reinterpret_cast<const v8h*>(base + 16 + lhalf * 8);
    return pack16(lo, hi);
}

// B fragment (32x16 f16): lane L<16 -> col L, K{0..7}+{16..23};
// lane L+16 -> col L, K{8..15}+{24..31}.
__device__ __forceinline__ v16h load_b_frag(const _Float16* base, int lhalf) {
    const _Float16* p = base + lhalf * 8;
    v8h lo = *reinterpret_cast<const v8h*>(p);
    v8h hi = *reinterpret_cast<const v8h*>(p + 16);
    return pack16(lo, hi);
}

// One-exp online softmax update in raw accumulator units (invT deferred):
//   cv  : raw feature dot (logit = cv * invT)
//   clv : label-cosine mask value (labels pre-normalized)
template <bool HAS_DIAG>
__device__ __forceinline__ void slot_update(float cv, float clv, bool isDiag, float invT,
                                            float& m, float& S, float& P, float& Ms) {
    const float d  = cv - m;                  // m starts at -inf: d=+inf, e=0, S=S*0+add
    const float e  = __expf(-fabsf(d) * invT);
    const bool  gt = d > 0.0f;
    float addGt = 1.0f, addLe = e, mv = clv;
    if (HAS_DIAG) {                           // diagonal: contributes to max only
        addGt = isDiag ? 0.0f : 1.0f;
        addLe = isDiag ? 0.0f : e;
        mv    = isDiag ? 0.0f : clv;
    }
    S = gt ? fmaf(S, e, addGt) : (S + addLe);
    m = gt ? cv : m;                          // reuse compare; no canonicalize max
    P  = fmaf(mv, cv, P);
    Ms += mv;
}

#define WAVES_PER_BLOCK 8

__global__ __launch_bounds__(32 * WAVES_PER_BLOCK) void supcon_main(
        const _Float16* __restrict__ F16, const _Float16* __restrict__ Lp,
        float* __restrict__ out, int n2) {
    constexpr int D  = 128;
    constexpr int KC = 4;                    // D / 32
    constexpr int W  = WAVES_PER_BLOCK;

    const int lane  = threadIdx.x & 31;
    // SGPR (wave-uniform) wave index: keeps the diagonal-tile branch scalar.
    const int waveU = __builtin_amdgcn_readfirstlane(threadIdx.x >> 5);
    const int l15   = lane & 15;
    const int lhalf = lane >> 4;
    const int nTiles = n2 >> 4;
    const int itile  = blockIdx.x;           // one i-tile per block, waves split j

    __shared__ float sm[W][16], sS[W][16], sP[W][16], sM[W][16];

    // Loop-invariant A fragments for this block's i-tile.
    const int arow = itile * 16 + l15;
    v16h aF[KC];
#pragma unroll
    for (int kc = 0; kc < KC; ++kc)
        aF[kc] = load_a_frag(F16 + (size_t)arow * D + kc * 32, lhalf);
    const v16h aL = load_a_frag(Lp + (size_t)arow * 32, lhalf);

    float m[8], S[8], P[8], Ms[8];
#pragma unroll
    for (int r = 0; r < 8; ++r) { m[r] = -INFINITY; S[r] = 0.f; P[r] = 0.f; Ms[r] = 0.f; }

    const float invT = 1.0f / SC_TEMP;
    const int tilesPerWave = nTiles / W;     // 512/8 = 64
    const int jt0 = waveU * tilesPerWave;    // SGPR
    const int mbase = itile * 16 + lhalf * 8;

    for (int jj = 0; jj < tilesPerWave; ++jj) {
        const int jt   = jt0 + jj;           // SGPR
        const int brow = jt * 16 + l15;      // global j for this lane's column N

        v16h bF[KC];
#pragma unroll
        for (int kc = 0; kc < KC; ++kc)
            bF[kc] = load_b_frag(F16 + (size_t)brow * D + kc * 32, lhalf);
        const v16h bL = load_b_frag(Lp + (size_t)brow * 32, lhalf);

        v8f c = {};
#pragma unroll
        for (int kc = 0; kc < KC; ++kc)
            c = __builtin_amdgcn_wmma_f32_16x16x32_f16(false, aF[kc], false, bF[kc],
                                                       (short)0, c, false, false);
        v8f cl = {};
        cl = __builtin_amdgcn_wmma_f32_16x16x32_f16(false, aL, false, bL,
                                                    (short)0, cl, false, false);

        if (jt == itile) {                   // scalar compare: tile contains the diagonal
#pragma unroll
            for (int r = 0; r < 8; ++r)
                slot_update<true>(c[r], cl[r], (mbase + r) == brow, invT,
                                  m[r], S[r], P[r], Ms[r]);
        } else {                             // fast path: no diagonal checks
#pragma unroll
            for (int r = 0; r < 8; ++r)
                slot_update<false>(c[r], cl[r], false, invT,
                                   m[r], S[r], P[r], Ms[r]);
        }
    }

    // Apply deferred temperature scale: logits = c * invT.
#pragma unroll
    for (int r = 0; r < 8; ++r) { m[r] *= invT; P[r] *= invT; }

    // Intra-wave merge across the 16 lanes of each half-wave, then stash in LDS.
#pragma unroll
    for (int r = 0; r < 8; ++r) {
        float mr = m[r], Sr = S[r], Pr = P[r], Mr = Ms[r];
#pragma unroll
        for (int off = 1; off < 16; off <<= 1) {
            float mo = __shfl_xor(mr, off, 16);
            float So = __shfl_xor(Sr, off, 16);
            float Po = __shfl_xor(Pr, off, 16);
            float Mo = __shfl_xor(Mr, off, 16);
            float mn = fmaxf(mr, mo);
            Sr = Sr * __expf(mr - mn) + So * __expf(mo - mn);
            mr = mn; Pr += Po; Mr += Mo;
        }
        if (l15 == 0) {
            const int row = lhalf * 8 + r;
            sm[waveU][row] = mr; sS[waveU][row] = Sr;
            sP[waveU][row] = Pr; sM[waveU][row] = Mr;
        }
    }
    __syncthreads();

    // Wave 0 merges the W partial results and accumulates the loss.
    if (waveU == 0) {
        const int row = lane & 15;
        float mr = -INFINITY, Sr = 0.f, Pr = 0.f, Mr = 0.f;
#pragma unroll
        for (int w = 0; w < W; ++w) {
            const float mw = sm[w][row], Sw = sS[w][row];
            const float mn = fmaxf(mr, mw);
            Sr = Sr * __expf(mr - mn) + Sw * __expf(mw - mn);
            mr = mn;
            Pr += sP[w][row];
            Mr += sM[w][row];
        }
        const float lse  = mr + __logf(Sr);
        const float mlpp = (Pr - Mr * lse) / (Mr + 1e-8f);
        float acc = (lane < 16) ? mlpp : 0.0f;   // lanes 16..31 are duplicates
#pragma unroll
        for (int off = 1; off < 32; off <<= 1)
            acc += __shfl_xor(acc, off, 32);
        if (lane == 0)
            atomicAdd(out, -acc / (float)n2);    // loss = -(T/Tb)*mean, T/Tb == 1
    }
}

extern "C" void kernel_launch(void* const* d_in, const int* in_sizes, int n_in,
                              void* d_out, int out_size, void* d_ws, size_t ws_size,
                              hipStream_t stream) {
    const float* f1 = (const float*)d_in[0];
    const float* f2 = (const float*)d_in[1];
    const float* l1 = (const float*)d_in[2];
    const float* l2 = (const float*)d_in[3];
    // d_in[4] (mapped_anchors) is unused by the reference.

    const int N  = in_sizes[4];
    const int D  = in_sizes[0] / N;   // 128
    const int C  = in_sizes[2] / N;   // 10
    const int n2 = 2 * N;             // 8192

    char* ws = (char*)d_ws;
    size_t off = 0;
    _Float16* F16 = (_Float16*)(ws + off);
    off += (size_t)n2 * D * sizeof(_Float16);
    off = (off + 255) & ~(size_t)255;
    _Float16* Lp = (_Float16*)(ws + off);

    float* out = (float*)d_out;

    zero_out_kernel<<<1, 1, 0, stream>>>(out);

    const int totF = n2 * D;
    prep_features<<<(totF + 255) / 256, 256, 0, stream>>>(f1, f2, F16, N, D);
    prep_labels<<<(n2 + 255) / 256, 256, 0, stream>>>(l1, l2, Lp, N, C);

    const int nTiles = n2 / 16;   // 512 blocks, 8 waves each split the j-range
    supcon_main<<<nTiles, 32 * WAVES_PER_BLOCK, 0, stream>>>(F16, Lp, out, n2);
}
